// Gemma2GQA_24902220382347
// MI455X (gfx1250) — matrix operand order repaired and checked by reading, not compiled, not run on hardware
//
#include <hip/hip_runtime.h>

typedef __bf16 bf16;
typedef __bf16 bf16x8 __attribute__((ext_vector_type(8)));
typedef __bf16 v16bf  __attribute__((ext_vector_type(16)));
typedef float  v8f    __attribute__((ext_vector_type(8)));
typedef int    v4i    __attribute__((ext_vector_type(4)));

#define B_    2
#define S_    2048
#define HID_  2304
#define H_    8
#define HKV_  4
#define D_    256
#define NQKV_ 4096          // H*D + 2*HKV*D
#define MTOK_ 4096          // B*S
#define NEG_  (-1000000000.0f)

#if defined(__HIP_DEVICE_COMPILE__) && __has_builtin(__builtin_amdgcn_global_load_async_to_lds_b128)
#define HAVE_ASYNC_LDS 1
#else
#define HAVE_ASYNC_LDS 0
#endif

#if HAVE_ASYNC_LDS
typedef __attribute__((address_space(1))) v4i* gv4i_p;
typedef __attribute__((address_space(3))) v4i* lv4i_p;
typedef __attribute__((address_space(1))) void* gvoid_p;
typedef __attribute__((address_space(3))) void* lvoid_p;
#endif

static __device__ __forceinline__ v16bf mk16(bf16x8 lo, bf16x8 hi) {
  return __builtin_shufflevector(lo, hi, 0,1,2,3,4,5,6,7,8,9,10,11,12,13,14,15);
}
static __device__ __forceinline__ v8f zero8() {
  v8f v;
#pragma unroll
  for (int i = 0; i < 8; ++i) v[i] = 0.0f;
  return v;
}
#define WMMA_BF16(a,b,c) \
  __builtin_amdgcn_wmma_f32_16x16x32_bf16(false,(a),false,(b),(short)0,(c),false,false)

// 16-byte global -> LDS copy; async (GLOBAL_LOAD_ASYNC_TO_LDS_B128, ASYNCcnt) when available.
static __device__ __forceinline__ void cp16(const bf16* g, bf16* l) {
#if HAVE_ASYNC_LDS
  __builtin_amdgcn_global_load_async_to_lds_b128(
      (gv4i_p)(gvoid_p)(g),
      (lv4i_p)(lvoid_p)(l), 0, 0);
#else
  *(bf16x8*)l = *(const bf16x8*)g;
#endif
}
static __device__ __forceinline__ void wait_async0() {
#if HAVE_ASYNC_LDS
  asm volatile("s_wait_asynccnt 0x0" ::: "memory");
#endif
}
static __device__ __forceinline__ void wait_async4() {
#if HAVE_ASYNC_LDS
  asm volatile("s_wait_asynccnt 0x4" ::: "memory");
#endif
}

// ---------------------------------------------------------------- converts
__global__ void k_cast_bf16(const float* __restrict__ src, bf16* __restrict__ dst, int n) {
  int i = blockIdx.x * blockDim.x + threadIdx.x;
  if (i < n) dst[i] = (bf16)src[i];
}

// WqkvT[n][k] = W(k, n) packed over {Wq | Wk | Wv}, n in [0,4096), k in [0,2304)
__global__ void k_pack_wqkvT(const float* __restrict__ Wq, const float* __restrict__ Wk,
                             const float* __restrict__ Wv, bf16* __restrict__ WT) {
  int id = blockIdx.x * blockDim.x + threadIdx.x;
  if (id >= NQKV_ * HID_) return;
  int n = id / HID_, k = id % HID_;
  float v;
  if (n < 2048)      v = Wq[(size_t)k * 2048 + n];
  else if (n < 3072) v = Wk[(size_t)k * 1024 + (n - 2048)];
  else               v = Wv[(size_t)k * 1024 + (n - 3072)];
  WT[id] = (bf16)v;
}

// WoT[n][k] = Wo[k][n], n in [0,2304), k in [0,2048)
__global__ void k_woT(const float* __restrict__ Wo, bf16* __restrict__ WoT) {
  int id = blockIdx.x * blockDim.x + threadIdx.x;
  if (id >= HID_ * 2048) return;
  int n = id / 2048, k = id % 2048;
  WoT[id] = (bf16)Wo[(size_t)k * HID_ + n];
}

// ---------------------------------------------------------------- GEMM (double-buffered)
// C[M,N] = A[M,K] * BT[N,K]^T ; A,BT bf16 row-major (contiguous along K).
// Block: 128x128 tile, 256 threads (8 waves); wave = 64x32 (4x2 WMMA tiles); BK=32.
template <bool OUT_F32>
__global__ __launch_bounds__(256) void k_gemm(const bf16* __restrict__ A,
                                              const bf16* __restrict__ BT,
                                              void* __restrict__ Cout, int K, int N) {
  __shared__ __align__(16) bf16 sA[2][128][32];
  __shared__ __align__(16) bf16 sB[2][128][32];
  const int m0 = blockIdx.y * 128, n0 = blockIdx.x * 128;
  const int tid = threadIdx.x, w = tid >> 5, lane = tid & 31;
  const int hl = lane >> 4, ln = lane & 15;
  const int wm = (w >> 2) * 64, wn = (w & 3) * 32;

  v8f acc[4][2];
#pragma unroll
  for (int i = 0; i < 4; ++i)
#pragma unroll
    for (int j = 0; j < 2; ++j) acc[i][j] = zero8();

  const int lr = tid >> 1, lc = (tid & 1) * 16;
  const bf16* ga = A  + (size_t)(m0 + lr) * K + lc;
  const bf16* gb = BT + (size_t)(n0 + lr) * K + lc;
  const int nk = K >> 5;  // K / 32

  auto stage = [&](int i, int buf) {
    const bf16* a = ga + i * 32;
    const bf16* b = gb + i * 32;
    cp16(a,     &sA[buf][lr][lc]);
    cp16(a + 8, &sA[buf][lr][lc + 8]);
    cp16(b,     &sB[buf][lr][lc]);
    cp16(b + 8, &sB[buf][lr][lc + 8]);
    if (i + 2 < nk) {  // speculative prefetch two tiles ahead -> global_prefetch_b8
      __builtin_prefetch(ga + (i + 2) * 32, 0, 1);
      __builtin_prefetch(gb + (i + 2) * 32, 0, 1);
    }
  };

  stage(0, 0);
  for (int i = 0; i < nk; ++i) {
    const int cur = i & 1;
    if (i + 1 < nk) {
      stage(i + 1, cur ^ 1);  // overlap next-tile staging with current compute
      wait_async4();          // first 4 async ops (current tile) retired
    } else {
      wait_async0();
    }
    __syncthreads();

    v16bf bfr[2];
#pragma unroll
    for (int nt = 0; nt < 2; ++nt) {
      const bf16* p = &sB[cur][wn + nt * 16 + ln][hl * 8];
      bfr[nt] = mk16(*(const bf16x8*)p, *(const bf16x8*)(p + 16));
    }
#pragma unroll
    for (int mt = 0; mt < 4; ++mt) {
      const bf16* p = &sA[cur][wm + mt * 16 + ln][0];
      v16bf af = mk16(*(const bf16x8*)(p + hl * 8), *(const bf16x8*)(p + 16 + hl * 8));
      acc[mt][0] = WMMA_BF16(af, bfr[0], acc[mt][0]);
      acc[mt][1] = WMMA_BF16(af, bfr[1], acc[mt][1]);
    }
    __syncthreads();  // done reading `cur` before it gets restaged at i+2
  }

#pragma unroll
  for (int mt = 0; mt < 4; ++mt)
#pragma unroll
    for (int nt = 0; nt < 2; ++nt)
#pragma unroll
      for (int rr = 0; rr < 8; ++rr) {
        int gm = m0 + wm + mt * 16 + rr + 8 * hl;
        int gn = n0 + wn + nt * 16 + ln;
        float v = acc[mt][nt][rr];
        if (OUT_F32) ((float*)Cout)[(size_t)gm * N + gn] = v;
        else         ((bf16*)Cout)[(size_t)gm * N + gn] = (bf16)v;
      }
}

// ---------------------------------------------------------------- RoPE + V transpose
__global__ __launch_bounds__(128) void k_rope(const bf16* __restrict__ QKV,
                                              const int* __restrict__ pos_ids,
                                              bf16* __restrict__ Qr, bf16* __restrict__ Kr,
                                              bf16* __restrict__ Vt) {
  const int bs = blockIdx.x;
  const int b = bs / S_, s = bs % S_;
  const int i = threadIdx.x;  // 0..127 == freq index
  const float pos = (float)pos_ids[bs];
  // inv_freq = base^(-i/128); ln(10000)/128 = 0.07195578415606394
  const float ang = pos * __expf(-(float)i * 0.07195578415606394f);
  float sn, cs;
  __sincosf(ang, &sn, &cs);
  const bf16* row = QKV + (size_t)bs * NQKV_;

#pragma unroll
  for (int hh = 0; hh < H_; ++hh) {
    float x0 = (float)row[hh * D_ + i];
    float x1 = (float)row[hh * D_ + i + 128];
    bf16* q = Qr + ((size_t)(b * H_ + hh) * S_ + s) * D_;
    q[i]       = (bf16)(x0 * cs - x1 * sn);
    q[i + 128] = (bf16)(x1 * cs + x0 * sn);
  }
#pragma unroll
  for (int kv = 0; kv < HKV_; ++kv) {
    const bf16* ksrc = row + 2048 + kv * D_;
    float x0 = (float)ksrc[i];
    float x1 = (float)ksrc[i + 128];
    bf16* kd = Kr + ((size_t)(b * HKV_ + kv) * S_ + s) * D_;
    kd[i]       = (bf16)(x0 * cs - x1 * sn);
    kd[i + 128] = (bf16)(x1 * cs + x0 * sn);
    const bf16* vsrc = row + 3072 + kv * D_;
    bf16* vd = Vt + (size_t)(b * HKV_ + kv) * D_ * S_ + s;
    vd[(size_t)i * S_]         = vsrc[i];
    vd[(size_t)(i + 128) * S_] = vsrc[i + 128];
  }
}

// ---------------------------------------------------------------- flash attention (GQA)
// Block: (qblock, head, batch); 128 threads = 4 waves; each wave owns 16 q rows.
__global__ __launch_bounds__(128) void k_attn(const bf16* __restrict__ Qr,
                                              const bf16* __restrict__ Kr,
                                              const bf16* __restrict__ Vt,
                                              bf16* __restrict__ Obuf) {
  __shared__ __align__(16) bf16 sK[32][256];   // keys x D slice
  __shared__ __align__(16) bf16 sV[256][32];   // D x keys (from Vt)
  __shared__ __align__(16) bf16 sP[4][16][32]; // per-wave probs
  const int b = blockIdx.z, hh = blockIdx.y, qb = blockIdx.x;
  const int kvh = hh >> 1;  // n_rep = 2
  const int tid = threadIdx.x, w = tid >> 5, lane = tid & 31;
  const int hl = lane >> 4, ln = lane & 15;
  const int q0 = qb * 64;
  const int qrow = q0 + w * 16;

  // Q fragments held in registers: 8 k-steps over D=256
  v16bf qf[8];
  const bf16* qp = Qr + ((size_t)(b * H_ + hh) * S_ + qrow + ln) * D_;
#pragma unroll
  for (int kk = 0; kk < 8; ++kk) {
    bf16x8 lo = *(const bf16x8*)(qp + kk * 32 + hl * 8);
    bf16x8 hi = *(const bf16x8*)(qp + kk * 32 + 16 + hl * 8);
    qf[kk] = mk16(lo, hi);
  }

  v8f o[16];
#pragma unroll
  for (int t = 0; t < 16; ++t) o[t] = zero8();
  float mrow[8], lrow[8];
#pragma unroll
  for (int r = 0; r < 8; ++r) { mrow[r] = -3.0e38f; lrow[r] = 0.0f; }

  const bf16* kb = Kr + (size_t)(b * HKV_ + kvh) * S_ * D_;
  const bf16* vb = Vt + (size_t)(b * HKV_ + kvh) * D_ * S_;

  const int kend = q0 + 64;
  for (int k0 = 0; k0 < kend; k0 += 32) {
    // stage K (32x256) and Vt (256x32) tiles (async -> LDS when available)
#pragma unroll
    for (int i = 0; i < 8; ++i) {
      int c = tid + 128 * i;
      int r = c >> 5, off = (c & 31) * 8;
      cp16(kb + (size_t)(k0 + r) * D_ + off, &sK[r][off]);
      int rv = c >> 2, offv = (c & 3) * 8;
      cp16(vb + (size_t)rv * S_ + k0 + offv, &sV[rv][offv]);
    }
    wait_async0();
    __syncthreads();

    // S = Q K^T : both key-fragments loaded, then WMMAs issued back-to-back
    v8f sacc[2] = { zero8(), zero8() };
#pragma unroll
    for (int kk = 0; kk < 8; ++kk) {
      v16bf kf[2];
#pragma unroll
      for (int nt = 0; nt < 2; ++nt) {
        const bf16* p = &sK[nt * 16 + ln][kk * 32 + hl * 8];
        kf[nt] = mk16(*(const bf16x8*)p, *(const bf16x8*)(p + 16));
      }
      sacc[0] = WMMA_BF16(qf[kk], kf[0], sacc[0]);
      sacc[1] = WMMA_BF16(qf[kk], kf[1], sacc[1]);
    }

    // softcap + causal mask + online softmax (row = r + 8*hl per lane-half)
#pragma unroll
    for (int r = 0; r < 8; ++r) {
      const int qg = qrow + r + 8 * hl;
      float x0 = 50.0f * tanhf(sacc[0][r] * (0.0625f / 50.0f));
      float x1 = 50.0f * tanhf(sacc[1][r] * (0.0625f / 50.0f));
      if (k0 + ln > qg)      x0 += NEG_;
      if (k0 + 16 + ln > qg) x1 += NEG_;
      float t = fmaxf(x0, x1);
#pragma unroll
      for (int off = 1; off < 16; off <<= 1) t = fmaxf(t, __shfl_xor(t, off, 32));
      float mn = fmaxf(mrow[r], t);
      float corr = __expf(mrow[r] - mn);
      mrow[r] = mn;
      float p0 = __expf(x0 - mn), p1 = __expf(x1 - mn);
      float rs = p0 + p1;
#pragma unroll
      for (int off = 1; off < 16; off <<= 1) rs += __shfl_xor(rs, off, 32);
      lrow[r] = lrow[r] * corr + rs;
#pragma unroll
      for (int t2 = 0; t2 < 16; ++t2) o[t2][r] *= corr;
      const int m = r + 8 * hl;
      sP[w][m][ln]      = (bf16)p0;
      sP[w][m][16 + ln] = (bf16)p1;
    }

    // O += P V : fragments preloaded in groups of 4 so WMMAs issue back-to-back
    {
      const bf16* pp = &sP[w][ln][0];
      v16bf pf = mk16(*(const bf16x8*)(pp + hl * 8), *(const bf16x8*)(pp + 16 + hl * 8));
#pragma unroll
      for (int g = 0; g < 4; ++g) {
        v16bf vf[4];
#pragma unroll
        for (int j = 0; j < 4; ++j) {
          const bf16* vp = &sV[(g * 4 + j) * 16 + ln][hl * 8];
          vf[j] = mk16(*(const bf16x8*)vp, *(const bf16x8*)(vp + 16));
        }
#pragma unroll
        for (int j = 0; j < 4; ++j)
          o[g * 4 + j] = WMMA_BF16(pf, vf[j], o[g * 4 + j]);
      }
    }
    __syncthreads();
  }

  // finalize: O /= l ; scatter into [B,S,H*D] bf16
#pragma unroll
  for (int r = 0; r < 8; ++r) lrow[r] = 1.0f / lrow[r];
  bf16* ob = Obuf + (size_t)(b * S_ + qrow) * (H_ * D_) + hh * D_;
#pragma unroll
  for (int nt = 0; nt < 16; ++nt)
#pragma unroll
    for (int r = 0; r < 8; ++r) {
      int m = r + 8 * hl;
      ob[(size_t)m * (H_ * D_) + nt * 16 + ln] = (bf16)(o[nt][r] * lrow[r]);
    }
}

// ---------------------------------------------------------------- launcher
extern "C" void kernel_launch(void* const* d_in, const int* in_sizes, int n_in,
                              void* d_out, int out_size, void* d_ws, size_t ws_size,
                              hipStream_t stream) {
  const float* hidden = (const float*)d_in[0];
  // d_in[1] = attention_mask (pure causal; reconstructed analytically)
  const int*   pos    = (const int*)d_in[2];
  const float* Wq     = (const float*)d_in[3];
  const float* Wk     = (const float*)d_in[4];
  const float* Wv     = (const float*)d_in[5];
  const float* Wo     = (const float*)d_in[6];
  (void)in_sizes; (void)n_in; (void)out_size; (void)ws_size;

  char* ws = (char*)d_ws;
  size_t off = 0;
  bf16* Xbf   = (bf16*)(ws + off); off += (size_t)MTOK_ * HID_  * 2;
  bf16* WqkvT = (bf16*)(ws + off); off += (size_t)NQKV_ * HID_  * 2;
  bf16* WoT   = (bf16*)(ws + off); off += (size_t)HID_  * 2048  * 2;
  bf16* QKV   = (bf16*)(ws + off); off += (size_t)MTOK_ * NQKV_ * 2;
  bf16* Qr    = (bf16*)(ws + off); off += (size_t)B_ * H_   * S_ * D_ * 2;
  bf16* Kr    = (bf16*)(ws + off); off += (size_t)B_ * HKV_ * S_ * D_ * 2;
  bf16* Vt    = (bf16*)(ws + off); off += (size_t)B_ * HKV_ * D_ * S_ * 2;
  bf16* Obuf  = (bf16*)(ws + off); off += (size_t)MTOK_ * H_ * D_ * 2;

  int n1 = MTOK_ * HID_;
  k_cast_bf16<<<(n1 + 255) / 256, 256, 0, stream>>>(hidden, Xbf, n1);
  int n2 = NQKV_ * HID_;
  k_pack_wqkvT<<<(n2 + 255) / 256, 256, 0, stream>>>(Wq, Wk, Wv, WqkvT);
  int n3 = HID_ * 2048;
  k_woT<<<(n3 + 255) / 256, 256, 0, stream>>>(Wo, WoT);

  // QKV = X * Wqkv : [4096,2304] x [2304,4096]
  k_gemm<false><<<dim3(NQKV_ / 128, MTOK_ / 128), 256, 0, stream>>>(Xbf, WqkvT, (void*)QKV, HID_, NQKV_);

  k_rope<<<B_ * S_, 128, 0, stream>>>(QKV, pos, Qr, Kr, Vt);

  k_attn<<<dim3(S_ / 64, H_, B_), 128, 0, stream>>>(Qr, Kr, Vt, Obuf);

  // out = O * Wo : [4096,2048] x [2048,2304] -> fp32
  k_gemm<true><<<dim3(HID_ / 128, MTOK_ / 128), 256, 0, stream>>>(Obuf, WoT, d_out, 2048, HID_);
}
